// EdgePredictionGNN_39608188403840
// MI455X (gfx1250) — matrix operand order repaired and checked by reading, not compiled, not run on hardware
//
#include <hip/hip_runtime.h>
#include <hip/hip_bf16.h>

typedef __attribute__((ext_vector_type(16))) _Float16 v16h;
typedef __attribute__((ext_vector_type(8)))  _Float16 v8h;
typedef __attribute__((ext_vector_type(8)))  float    v8f;

#define NN 50000
#define NE 1000000
#define C  64
#define NEG_SLOPE 0.2f

// ---------- order-preserving float<->uint encoding for atomic segment-max ----------
__device__ __forceinline__ unsigned fenc(float f) {
  unsigned u = __float_as_uint(f);
  return (u & 0x80000000u) ? ~u : (u | 0x80000000u);
}
__device__ __forceinline__ float fdec(unsigned u) {
  u = (u & 0x80000000u) ? (u & 0x7FFFFFFFu) : ~u;
  return __uint_as_float(u);
}

// ---------- WMMA A fragment from an f16 row (CDNA5 ISA 7.12.2, f16 16x32) ----------
// lane L holds row M = L&15:
//   halfs 0..7  -> K = kbase + (L>=16 ? 8 : 0) + i        (16B contiguous)
//   halfs 8..15 -> K = kbase + 16 + (L>=16 ? 8 : 0) + i   (16B contiguous)
// rows are 128B aligned, kbase/kh multiples of 8 halfs -> both loads 16B aligned.
__device__ __forceinline__ v16h build_a_f16(const _Float16* __restrict__ row, int kbase, int lane) {
  const int kh = (lane & 16) ? 8 : 0;
  v8h lo = *(const v8h*)(row + kbase + kh);
  v8h hi = *(const v8h*)(row + kbase + 16 + kh);
  return __builtin_shufflevector(lo, hi, 0, 1, 2, 3, 4, 5, 6, 7,
                                 8, 9, 10, 11, 12, 13, 14, 15);
}

// ---------- one-time pack: W[K=64][64] f32 -> B fragments in register layout ----------
// B (32x16): lane L holds col N = L&15; halfs i -> K = kbase + (L>=16 ? 8 : 0) + i + (i>=8 ? 8 : 0)
// P layout: [(kt*4+nt)*32 + lane] * 16 halfs  (each lane's v16h contiguous, 32B)
__global__ void pack_b64(const float* __restrict__ W, _Float16* __restrict__ P) {
  const int t = blockIdx.x * blockDim.x + threadIdx.x;   // 8 frags * 32 lanes = 256
  if (t >= 256) return;
  const int lane = t & 31;
  const int frag = t >> 5;                 // 0..7
  const int kt = frag >> 2, nt = frag & 3;
  const int n = nt * 16 + (lane & 15);
  const int kh = (lane & 16) ? 8 : 0;
  _Float16* o = P + (size_t)t * 16;
  #pragma unroll
  for (int i = 0; i < 16; ++i)
    o[i] = (_Float16)W[(size_t)(kt * 32 + kh + i + (i >= 8 ? 8 : 0)) * C + n];
}

// ---------- f32 -> f16 array convert (for x) ----------
__global__ void to_f16(const float* __restrict__ A, _Float16* __restrict__ B, int nelem) {
  const int t = blockIdx.x * blockDim.x + threadIdx.x;
  if (t < nelem) B[t] = (_Float16)A[t];
}

// ---------- kernel 1: H[nrows,64] = X16[nrows,64] @ W (packed frags) via WMMA ----------
__global__ void node_gemm_wmma(const _Float16* __restrict__ X16,
                               const _Float16* __restrict__ Wp,
                               float* __restrict__ H, int nrows) {
  const int lane = threadIdx.x & 31;
  const int tile = blockIdx.x * (blockDim.x >> 5) + (threadIdx.x >> 5);
  const int row0 = tile * 16;
  if (row0 >= nrows) return;               // wave-uniform: EXEC stays all-ones for WMMA
  const int m = lane & 15;
  const v16h* __restrict__ bf = (const v16h*)Wp;

  v8f acc[4];
  #pragma unroll
  for (int nt = 0; nt < 4; ++nt)
    #pragma unroll
    for (int r = 0; r < 8; ++r) acc[nt][r] = 0.f;

  #pragma unroll
  for (int kt = 0; kt < 2; ++kt) {
    v16h a = build_a_f16(X16 + (size_t)(row0 + m) * C, kt * 32, lane);
    #pragma unroll
    for (int nt = 0; nt < 4; ++nt) {
      v16h b = bf[(kt * 4 + nt) * 32 + lane];      // 32B contiguous per lane
      acc[nt] = __builtin_amdgcn_wmma_f32_16x16x32_f16(false, a, false, b,
                                                       (short)0, acc[nt], false, false);
    }
  }
  const int mo = (lane & 16) ? 8 : 0;      // C/D layout: VGPR r -> row r (+8 for hi half)
  #pragma unroll
  for (int nt = 0; nt < 4; ++nt)
    #pragma unroll
    for (int r = 0; r < 8; ++r)
      H[(size_t)(row0 + r + mo) * C + nt * 16 + m] = acc[nt][r];
}

// ---------- kernel 2: per-node attention dots + reset segment state ----------
__global__ void node_prep(const float* __restrict__ H, const float* __restrict__ att_l,
                          const float* __restrict__ att_r, float* __restrict__ al,
                          float* __restrict__ ar, unsigned* __restrict__ maxb,
                          float* __restrict__ sumb, float* __restrict__ hacc) {
  const int n = blockIdx.x * blockDim.x + threadIdx.x;
  if (n >= NN) return;
  const float4* row = (const float4*)(H + (size_t)n * C);
  float4* z = (float4*)(hacc + (size_t)n * C);
  float sl = 0.f, sr = 0.f;
  #pragma unroll
  for (int q = 0; q < C / 4; ++q) {
    const float4 v = row[q];
    const float4 l = ((const float4*)att_l)[q];
    const float4 r = ((const float4*)att_r)[q];
    sl += v.x * l.x + v.y * l.y + v.z * l.z + v.w * l.w;
    sr += v.x * r.x + v.y * r.y + v.z * r.z + v.w * r.w;
    z[q] = make_float4(0.f, 0.f, 0.f, 0.f);
  }
  al[n] = sl; ar[n] = sr;
  maxb[n] = 0x007FFFFFu;                   // fenc(-inf)
  sumb[n] = 0.f;
}

// ---------- kernel 3: raw attention logit per edge + atomic segment max ----------
__global__ void edge_alpha(const int* __restrict__ src, const int* __restrict__ dst,
                           const float* __restrict__ H, const float* __restrict__ al,
                           const float* __restrict__ ar, float* __restrict__ alpha,
                           unsigned* __restrict__ maxb, int ne_total) {
  const int e = blockIdx.x * blockDim.x + threadIdx.x;
  if (e >= ne_total) return;
  int j, i;
  if (e < NE) { j = src[e]; i = dst[e]; } else { j = i = e - NE; }   // self-loops appended
  const float4* hj = (const float4*)(H + (size_t)j * C);
  const float4* hi = (const float4*)(H + (size_t)i * C);
  float d = 0.f;
  #pragma unroll
  for (int q = 0; q < C / 4; ++q) {
    const float4 a4 = hi[q], b4 = hj[q];
    d += a4.x * b4.x + a4.y * b4.y + a4.z * b4.z + a4.w * b4.w;      // MX gating logit
  }
  float a = al[j] + ar[i];
  a *= 1.f / (1.f + __expf(-d));                                     // sigmoid gate
  a = (a > 0.f) ? a : NEG_SLOPE * a;                                 // leaky relu
  alpha[e] = a;
  atomicMax(&maxb[i], fenc(a));
}

// ---------- kernel 4: exp(alpha - max) + atomic segment sum ----------
__global__ void edge_expsum(const int* __restrict__ dst, float* __restrict__ alpha,
                            const unsigned* __restrict__ maxb, float* __restrict__ sumb,
                            int ne_total) {
  const int e = blockIdx.x * blockDim.x + threadIdx.x;
  if (e >= ne_total) return;
  const int i = (e < NE) ? dst[e] : e - NE;
  const float ex = __expf(alpha[e] - fdec(maxb[i]));
  alpha[e] = ex;
  atomicAdd(&sumb[i], ex);
}

// ---------- kernel 5: normalized weighted scatter (4 threads / edge) ----------
__global__ void edge_scatter(const int* __restrict__ src, const int* __restrict__ dst,
                             const float* __restrict__ H, const float* __restrict__ alpha,
                             const float* __restrict__ sumb, float* __restrict__ hacc,
                             int ne_total) {
  const int t = blockIdx.x * blockDim.x + threadIdx.x;
  const int e = t >> 2;
  if (e >= ne_total) return;
  const int f0 = (t & 3) * 16;
  int j, i;
  if (e < NE) { j = src[e]; i = dst[e]; } else { j = i = e - NE; }
  const float coef = alpha[e] / sumb[i];
  const float4* hj = (const float4*)(H + (size_t)j * C + f0);
  float* o = hacc + (size_t)i * C + f0;
  #pragma unroll
  for (int q = 0; q < 4; ++q) {
    const float4 v = hj[q];
    atomicAdd(&o[q * 4 + 0], v.x * coef);
    atomicAdd(&o[q * 4 + 1], v.y * coef);
    atomicAdd(&o[q * 4 + 2], v.z * coef);
    atomicAdd(&o[q * 4 + 3], v.w * coef);
  }
}

// ---------- kernel 6: bias + relu, emit f32 + f16 shadow ----------
__global__ void node_finish(const float* __restrict__ hacc, const float* __restrict__ bias,
                            float* __restrict__ Hout, _Float16* __restrict__ Hout16) {
  const int t = blockIdx.x * blockDim.x + threadIdx.x;
  if (t >= NN * C) return;
  float v = hacc[t] + bias[t & (C - 1)];
  v = fmaxf(v, 0.f);
  Hout[t] = v;
  Hout16[t] = (_Float16)v;
}

// ---------- kernel 7: edge MLP. out[e] = relu([h_src|ea|h_dst] @ Wm1 + bm1) @ Wm2 + bm2
// one wave = 16 edges; K=129 split: src(K=64) GEMM + dst(K=64) GEMM + rank-1 ea term
__global__ void edge_mlp_wmma(const _Float16* __restrict__ H16,
                              const int* __restrict__ src, const int* __restrict__ dst,
                              const float* __restrict__ eattr,
                              const _Float16* __restrict__ Wm1sp,   // packed rows 0..63
                              const _Float16* __restrict__ Wm1dp,   // packed rows 65..128
                              const float* __restrict__ Wm1,        // for row 64
                              const float* __restrict__ bm1,
                              const float* __restrict__ Wm2, const float* __restrict__ bm2,
                              float* __restrict__ out) {
  const int lane = threadIdx.x & 31;
  const int tile = blockIdx.x * (blockDim.x >> 5) + (threadIdx.x >> 5);
  const int eb = tile * 16;
  if (eb >= NE) return;                    // wave-uniform
  const int m  = lane & 15;
  const int mo = (lane & 16) ? 8 : 0;
  const v16h* __restrict__ bfs = (const v16h*)Wm1sp;
  const v16h* __restrict__ bfd = (const v16h*)Wm1dp;

  // accumulator init: edge_attr rank-1 term (Wm1 row 64) + bias bm1
  float ea[8];
  #pragma unroll
  for (int r = 0; r < 8; ++r) ea[r] = eattr[eb + r + mo];
  v8f acc[4];
  #pragma unroll
  for (int nt = 0; nt < 4; ++nt) {
    const int n = nt * 16 + m;
    const float wmid = Wm1[(size_t)64 * C + n];
    const float bb = bm1[n];
    #pragma unroll
    for (int r = 0; r < 8; ++r) acc[nt][r] = ea[r] * wmid + bb;
  }

  const int js = src[eb + m];              // gathered A rows (L2-resident f16 table)
  const int jd = dst[eb + m];

  #pragma unroll
  for (int kt = 0; kt < 2; ++kt) {         // src_h @ Wm1[0:64]
    v16h a = build_a_f16(H16 + (size_t)js * C, kt * 32, lane);
    #pragma unroll
    for (int nt = 0; nt < 4; ++nt) {
      v16h b = bfs[(kt * 4 + nt) * 32 + lane];
      acc[nt] = __builtin_amdgcn_wmma_f32_16x16x32_f16(false, a, false, b,
                                                       (short)0, acc[nt], false, false);
    }
  }
  #pragma unroll
  for (int kt = 0; kt < 2; ++kt) {         // dst_h @ Wm1[65:129]
    v16h a = build_a_f16(H16 + (size_t)jd * C, kt * 32, lane);
    #pragma unroll
    for (int nt = 0; nt < 4; ++nt) {
      v16h b = bfd[(kt * 4 + nt) * 32 + lane];
      acc[nt] = __builtin_amdgcn_wmma_f32_16x16x32_f16(false, a, false, b,
                                                       (short)0, acc[nt], false, false);
    }
  }

  // relu, then reduce over hidden dim with Wm2.
  // C/D layout: lane = N (within 16), so summing over N = xor-reduce across each 16-lane half.
  float res[8];
  #pragma unroll
  for (int r = 0; r < 8; ++r) res[r] = 0.f;
  #pragma unroll
  for (int nt = 0; nt < 4; ++nt) {
    const float w2 = Wm2[nt * 16 + m];
    #pragma unroll
    for (int r = 0; r < 8; ++r) {
      float v = fmaxf(acc[nt][r], 0.f) * w2;
      v += __shfl_xor(v, 1);
      v += __shfl_xor(v, 2);
      v += __shfl_xor(v, 4);
      v += __shfl_xor(v, 8);
      res[r] += v;
    }
  }
  if (m == 0) {                            // lanes 0 and 16 hold the two M-halves
    const float b2 = bm2[0];
    #pragma unroll
    for (int r = 0; r < 8; ++r) out[eb + r + mo] = res[r] + b2;
  }
}

extern "C" void kernel_launch(void* const* d_in, const int* in_sizes, int n_in,
                              void* d_out, int out_size, void* d_ws, size_t ws_size,
                              hipStream_t stream) {
  const float* x      = (const float*)d_in[0];
  const int*   ei     = (const int*)d_in[1];
  const float* eattr  = (const float*)d_in[2];
  const float* W1     = (const float*)d_in[3];
  const float* att_l1 = (const float*)d_in[4];
  const float* att_r1 = (const float*)d_in[5];
  const float* b1     = (const float*)d_in[6];
  const float* W2     = (const float*)d_in[7];
  const float* att_l2 = (const float*)d_in[8];
  const float* att_r2 = (const float*)d_in[9];
  const float* b2     = (const float*)d_in[10];
  const float* Wm1    = (const float*)d_in[11];
  const float* bm1    = (const float*)d_in[12];
  const float* Wm2    = (const float*)d_in[13];
  const float* bm2    = (const float*)d_in[14];
  float* out = (float*)d_out;

  const int* esrc = ei;
  const int* edst = ei + NE;

  // --- carve workspace ---
  char* p = (char*)d_ws;
  auto carve = [&](size_t bytes) -> void* {
    void* r = (void*)p;
    p += (bytes + 255) & ~(size_t)255;
    return r;
  };
  float*    htmp  = (float*)carve(sizeof(float) * (size_t)NN * C);   // GEMM out / attn input
  float*    h1    = (float*)carve(sizeof(float) * (size_t)NN * C);
  float*    h2    = (float*)carve(sizeof(float) * (size_t)NN * C);
  _Float16* g16a  = (_Float16*)carve(sizeof(_Float16) * (size_t)NN * C); // x16 then h1_16
  _Float16* g16b  = (_Float16*)carve(sizeof(_Float16) * (size_t)NN * C); // h2_16
  _Float16* W1p   = (_Float16*)carve(sizeof(_Float16) * 4096);       // packed B frags
  _Float16* W2p   = (_Float16*)carve(sizeof(_Float16) * 4096);
  _Float16* Wm1sp = (_Float16*)carve(sizeof(_Float16) * 4096);
  _Float16* Wm1dp = (_Float16*)carve(sizeof(_Float16) * 4096);
  float*    al    = (float*)carve(sizeof(float) * NN);
  float*    ar    = (float*)carve(sizeof(float) * NN);
  unsigned* maxb  = (unsigned*)carve(sizeof(unsigned) * NN);
  float*    sumb  = (float*)carve(sizeof(float) * NN);
  float*    alphw = (float*)carve(sizeof(float) * (size_t)(NE + NN));

  const int EA = NE + NN;
  const int gemmBlocks = ((NN / 16) + 3) / 4;          // 4 waves / block
  const int nodeBlocks = (NN + 255) / 256;
  const int edgeBlocks = (EA + 255) / 256;
  const int scatBlocks = (EA * 4 + 255) / 256;
  const int elemBlocks = (NN * C + 255) / 256;
  const int mlpBlocks  = (NE / 16 + 3) / 4;

  // ---- one-time packing ----
  pack_b64<<<1, 256, 0, stream>>>(W1, W1p);
  pack_b64<<<1, 256, 0, stream>>>(W2, W2p);
  pack_b64<<<1, 256, 0, stream>>>(Wm1, Wm1sp);                 // rows 0..63
  pack_b64<<<1, 256, 0, stream>>>(Wm1 + (size_t)65 * C, Wm1dp); // rows 65..128
  to_f16<<<elemBlocks, 256, 0, stream>>>(x, g16a, NN * C);

  // ---- layer 1 ----
  node_gemm_wmma<<<gemmBlocks, 128, 0, stream>>>(g16a, W1p, htmp, NN);
  node_prep<<<nodeBlocks, 256, 0, stream>>>(htmp, att_l1, att_r1, al, ar, maxb, sumb, h1);
  edge_alpha<<<edgeBlocks, 256, 0, stream>>>(esrc, edst, htmp, al, ar, alphw, maxb, EA);
  edge_expsum<<<edgeBlocks, 256, 0, stream>>>(edst, alphw, maxb, sumb, EA);
  edge_scatter<<<scatBlocks, 256, 0, stream>>>(esrc, edst, htmp, alphw, sumb, h1, EA);
  node_finish<<<elemBlocks, 256, 0, stream>>>(h1, b1, h1, g16a);  // g16a now h1_16

  // ---- layer 2 ----
  node_gemm_wmma<<<gemmBlocks, 128, 0, stream>>>(g16a, W2p, htmp, NN);
  node_prep<<<nodeBlocks, 256, 0, stream>>>(htmp, att_l2, att_r2, al, ar, maxb, sumb, h2);
  edge_alpha<<<edgeBlocks, 256, 0, stream>>>(esrc, edst, htmp, al, ar, alphw, maxb, EA);
  edge_expsum<<<edgeBlocks, 256, 0, stream>>>(edst, alphw, maxb, sumb, EA);
  edge_scatter<<<scatBlocks, 256, 0, stream>>>(esrc, edst, htmp, alphw, sumb, h2, EA);
  node_finish<<<elemBlocks, 256, 0, stream>>>(h2, b2, h2, g16b);  // g16b = h2_16

  // ---- edge MLP (16.6 GFLOP — WMMA path) ----
  edge_mlp_wmma<<<mlpBlocks, 128, 0, stream>>>(g16b, esrc, edst, eattr,
                                               Wm1sp, Wm1dp, Wm1, bm1, Wm2, bm2, out);
}
